// VisionSdpaAttentionCausal_26860725469791
// MI455X (gfx1250) — matrix operand order repaired and checked by reading, not compiled, not run on hardware
//
#include <hip/hip_runtime.h>
#include <stdint.h>
#include <stddef.h>

// Problem constants (reference: L=1024, B=8, C=1024, H=16, D=64)
#define LQ   1024
#define BB   8
#define CC   1024
#define HH   16
#define DD   64
#define MTOT (LQ * BB)   // 8192 rows (l major, b minor)
#define NQKV (3 * CC)    // 3072
#define KD   CC          // 1024 reduction dim for both GEMMs

typedef __bf16 bf16;
typedef bf16  v16bf __attribute__((ext_vector_type(16)));
typedef bf16  v8bf  __attribute__((ext_vector_type(8)));
typedef float v8f   __attribute__((ext_vector_type(8)));

union Frag { v16bf v; v8bf h[2]; };

static __device__ inline bf16 f2bf(float f) {
  unsigned u = __float_as_uint(f);
  unsigned r = u + 0x7fffu + ((u >> 16) & 1u);   // round-to-nearest-even
  unsigned short s = (unsigned short)(r >> 16);
  return __builtin_bit_cast(bf16, s);
}

static __device__ inline v8f zero8() {
  v8f z;
#pragma unroll
  for (int i = 0; i < 8; ++i) z[i] = 0.f;
  return z;
}

// CDNA5 async global->LDS copy (ASYNCcnt path), GVS addressing:
// mem = SADDR(64b sgpr pair) + VADDR(32b vgpr offset); VDST = per-lane LDS byte addr.
static __device__ inline void async_b128(unsigned lds_addr, unsigned voff, unsigned long long sbase) {
  asm volatile("global_load_async_to_lds_b128 %0, %1, %2"
               :: "v"(lds_addr), "v"(voff), "s"(sbase) : "memory");
}
// Wait until this wave's outstanding async ops <= N (in-order completion:
// allows the next k-step's 4 in-flight copies to stay outstanding).
static __device__ inline void wait_async0() {
  asm volatile("s_wait_asynccnt 0x0" ::: "memory");
}
static __device__ inline void wait_async4() {
  asm volatile("s_wait_asynccnt 0x4" ::: "memory");
}

static __device__ inline v8f wmma_bf16(v16bf a, v16bf b, v8f c) {
  return __builtin_amdgcn_wmma_f32_16x16x32_bf16(false, a, false, b, (short)0, c, false, false);
}

// ---------------------------------------------------------------- cvt f32->bf16
__global__ void cvt_f32_bf16_kernel(const float* __restrict__ x, bf16* __restrict__ y, int n) {
  int i = blockIdx.x * 256 + threadIdx.x;
  if (i < n) y[i] = f2bf(x[i]);
}

// ---------------------------------------------------------------- GEMM tiling
#define BM  128
#define BN  128
#define BK  32
#define LDH 40                      // padded LDS row stride in bf16 (80B)
#define BUFH (BM * LDH)             // halves per LDS buffer
#define BUFB (BUFH * 2)             // bytes per LDS buffer (10240)

// Stage a BM x BK (and BN x BK) bf16 block into an LDS buffer with async b128
// copies. Each of 256 threads moves 2 x 16B chunks per matrix -> 4 async ops/wave-slot.
static __device__ inline void stage_block(unsigned ldsA, unsigned ldsB,
                                          unsigned long long gA, unsigned long long gB,
                                          int kb, int tid) {
#pragma unroll
  for (int i = 0; i < 2; ++i) {
    int c  = tid + 256 * i;       // 0..511 chunks of 16B (rows of 64B, 4 chunks/row)
    int r  = c >> 2;              // row 0..127
    int cb = (c & 3) * 16;        // byte offset within 64B row
    unsigned voff = (unsigned)(r * (KD * 2) + kb * 2 + cb);
    unsigned loff = (unsigned)(r * (LDH * 2) + cb);
    async_b128(ldsA + loff, voff, gA);
    async_b128(ldsB + loff, voff, gB);
  }
}

// Load A-fragment (16x32 bf16) from LDS: lane=row, K = {klo..klo+7, klo+16..klo+23}
static __device__ inline v16bf load_fragA(const bf16* base, int row, int klo) {
  Frag f;
  const bf16* p = base + row * LDH + klo;
  f.h[0] = *(const v8bf*)p;
  f.h[1] = *(const v8bf*)(p + 16);
  return f.v;
}
// Load B-fragment (32x16 bf16) from LDS [n][k] tile: lane=col n, K = {kbb..kbb+7, kbb+16..kbb+23}
static __device__ inline v16bf load_fragB(const bf16* base, int nrow, int kbb) {
  Frag f;
  const bf16* p = base + nrow * LDH + kbb;
  f.h[0] = *(const v8bf*)p;
  f.h[1] = *(const v8bf*)(p + 16);
  return f.v;
}

// Shared double-buffered WMMA mainloop: acc[2][4] over a 32x64 wave tile.
static __device__ inline void gemm_mainloop(const bf16* As, const bf16* Bs,
                                            unsigned ldsA, unsigned ldsB,
                                            unsigned long long gA, unsigned long long gB,
                                            int wm, int wn, int ln, int hi, int tid,
                                            v8f acc[2][4]) {
  const int klo = hi * 8;
  const int kbb = hi * 8;
  stage_block(ldsA, ldsB, gA, gB, 0, tid);           // prologue: buffer 0
  int cur = 0;
  for (int kb = 0; kb < KD; kb += BK) {
    const bool more = (kb + BK) < KD;
    if (more) stage_block(ldsA + (cur ^ 1) * BUFB, ldsB + (cur ^ 1) * BUFB,
                          gA, gB, kb + BK, tid);     // issue next buffer first
    if (more) wait_async4(); else wait_async0();     // current buffer resident
    __syncthreads();
    const bf16* Ab = As + cur * BUFH;
    const bf16* Bb = Bs + cur * BUFH;
    Frag aF[2];
#pragma unroll
    for (int mt = 0; mt < 2; ++mt) aF[mt].v = load_fragA(Ab, wm * 32 + mt * 16 + ln, klo);
#pragma unroll
    for (int nt = 0; nt < 4; ++nt) {
      v16bf bF = load_fragB(Bb, wn * 64 + nt * 16 + ln, kbb);
#pragma unroll
      for (int mt = 0; mt < 2; ++mt) acc[mt][nt] = wmma_bf16(aF[mt].v, bF, acc[mt][nt]);
    }
    __syncthreads();                                 // all waves done before overwrite
    cur ^= 1;
  }
}

// ---------------------------------------------------------------- QKV GEMM + bias + RoPE
// C[m][n] = sum_k hs_bf16[m][k] * Wqkv[n][k]  (m=(l,b) l-major; n = which*1024 + h*64 + d)
__global__ void __launch_bounds__(256)
qkv_rope_kernel(const bf16* __restrict__ A, const bf16* __restrict__ W,
                const float* __restrict__ bias, const float* __restrict__ rpe,
                bf16* __restrict__ Qb, bf16* __restrict__ Kb, bf16* __restrict__ Vt) {
  __shared__ __align__(16) bf16 As[2 * BUFH];
  __shared__ __align__(16) bf16 Bs[2 * BUFH];
  const int tid  = threadIdx.x;
  const int lane = tid & 31;
  const int wid  = tid >> 5;
  const int wm = wid & 3, wn = wid >> 2;     // 4 x 2 wave grid
  const int m0 = blockIdx.y * BM;
  const int n0 = blockIdx.x * BN;
  const int ln = lane & 15;
  const int hi = lane >> 4;

  const unsigned ldsA = (unsigned)(size_t)As;
  const unsigned ldsB = (unsigned)(size_t)Bs;
  const unsigned long long gA = (unsigned long long)(size_t)(A + (size_t)m0 * KD);
  const unsigned long long gB = (unsigned long long)(size_t)(W + (size_t)n0 * KD);

  v8f acc[2][4];
#pragma unroll
  for (int i = 0; i < 2; ++i)
#pragma unroll
    for (int j = 0; j < 4; ++j) acc[i][j] = zero8();

  gemm_mainloop(As, Bs, ldsA, ldsB, gA, gB, wm, wn, ln, hi, tid, acc);

  // Epilogue: +bias, RoPE on q/k (pair exchange n<->n^1 == lane^1), scatter to attention layouts.
  const int rsel = hi * 8;
#pragma unroll
  for (int mt = 0; mt < 2; ++mt) {
#pragma unroll
    for (int nt = 0; nt < 4; ++nt) {
      int n = n0 + wn * 64 + nt * 16 + ln;
      int which = n >> 10;           // 0=q 1=k 2=v (wave-uniform within nt)
      int rem = n & 1023;
      int h = rem >> 6, d = rem & 63;
      float bv = bias[n];
#pragma unroll
      for (int r = 0; r < 8; ++r) {
        int m = m0 + wm * 32 + mt * 16 + rsel + r;
        int l = m >> 3;              // m = l*8 + b
        int b = m & 7;
        float val = acc[mt][nt][r] + bv;
        if (which < 2) {
          float prt = __shfl_xor(val, 1, 32);
          float rot = (d & 1) ? prt : -prt;
          float ang = rpe[l * 32 + (d & 31)];
          float sn, cs;
          __sincosf(ang, &sn, &cs);
          val = val * cs + rot * sn;
        }
        bf16 o = f2bf(val);
        size_t head = (size_t)b * HH + h;
        if (which == 0)      Qb[(head * LQ + l) * DD + d] = o;
        else if (which == 1) Kb[(head * LQ + l) * DD + d] = o;
        else                 Vt[(head * DD + d) * LQ + l] = o;   // V transposed (b,h,d,l)
      }
    }
  }
}

// ---------------------------------------------------------------- flash attention
// One wave = 16 queries of one head; online softmax over 32-key blocks.
__global__ void __launch_bounds__(128)
attn_kernel(const bf16* __restrict__ Qb, const bf16* __restrict__ Kb,
            const bf16* __restrict__ Vt, const int* __restrict__ mask,
            bf16* __restrict__ Ob) {
  __shared__ __align__(16) bf16 Pt[4][16 * LDH];   // per-wave P staging
  const int tid  = threadIdx.x;
  const int lane = tid & 31;
  const int w    = tid >> 5;
  const int bh = blockIdx.x >> 4;                  // 0..127 head index (b*16+h)
  const int qc = blockIdx.x & 15;                  // 64-query chunk
  const int q0 = qc * 64 + w * 16;
  const int b  = bh >> 4;
  const int h  = bh & 15;
  const int ln = lane & 15;
  const int hi = lane >> 4;

  const bf16* Qh = Qb + (size_t)bh * LQ * DD;
  const bf16* Kh = Kb + (size_t)bh * LQ * DD;
  const bf16* Vh = Vt + (size_t)bh * DD * LQ;

  // Q A-fragments for both 32-wide d-steps (kept in registers)
  Frag qF[2];
  {
    const bf16* pq = Qh + (size_t)(q0 + ln) * DD;
    int klo = hi * 8;
#pragma unroll
    for (int s = 0; s < 2; ++s) {
      qF[s].h[0] = *(const v8bf*)(pq + s * 32 + klo);
      qF[s].h[1] = *(const v8bf*)(pq + s * 32 + klo + 16);
    }
  }

  v8f O[4];
#pragma unroll
  for (int dt = 0; dt < 4; ++dt) O[dt] = zero8();
  float m_i[8], l_i[8];
#pragma unroll
  for (int r = 0; r < 8; ++r) { m_i[r] = -3.0e38f; l_i[r] = 0.f; }
  const float scale = 0.125f;   // 1/sqrt(64)

  for (int kb = 0; kb < LQ; kb += 32) {
    // prefetch next key block of K and V into cache (global_prefetch)
    if (kb + 32 < LQ) {
      __builtin_prefetch((const void*)(Kh + (size_t)(kb + 32 + lane) * DD), 0, 3);
      __builtin_prefetch((const void*)(Vh + (size_t)(lane * 2) * LQ + kb + 32), 0, 3);
    }
    // S = Q @ K^T for two 16-key tiles
    v8f S[2];
#pragma unroll
    for (int kt = 0; kt < 2; ++kt) {
      v8f s = zero8();
      const bf16* pk = Kh + (size_t)(kb + kt * 16 + ln) * DD + hi * 8;
#pragma unroll
      for (int st = 0; st < 2; ++st) {
        Frag kF;
        kF.h[0] = *(const v8bf*)(pk + st * 32);
        kF.h[1] = *(const v8bf*)(pk + st * 32 + 16);
        s = wmma_bf16(qF[st].v, kF.v, s);
      }
      S[kt] = s;
    }
    // mask + scale + online softmax update
    float p0[8], p1[8];
#pragma unroll
    for (int r = 0; r < 8; ++r) {
      int q = q0 + hi * 8 + r;
      const int* mrow = mask + ((size_t)b * LQ + q) * LQ + kb;
      p0[r] = mrow[ln]      ? S[0][r] * scale : -3.0e38f;
      p1[r] = mrow[16 + ln] ? S[1][r] * scale : -3.0e38f;
    }
#pragma unroll
    for (int r = 0; r < 8; ++r) {
      float rm = fmaxf(p0[r], p1[r]);
      rm = fmaxf(rm, __shfl_xor(rm, 1, 32));
      rm = fmaxf(rm, __shfl_xor(rm, 2, 32));
      rm = fmaxf(rm, __shfl_xor(rm, 4, 32));
      rm = fmaxf(rm, __shfl_xor(rm, 8, 32));
      float mn = fmaxf(m_i[r], rm);
      float alpha = __expf(m_i[r] - mn);
      float e0 = __expf(p0[r] - mn);
      float e1 = __expf(p1[r] - mn);
      float rs = e0 + e1;
      rs += __shfl_xor(rs, 1, 32);
      rs += __shfl_xor(rs, 2, 32);
      rs += __shfl_xor(rs, 4, 32);
      rs += __shfl_xor(rs, 8, 32);
      l_i[r] = l_i[r] * alpha + rs;
      m_i[r] = mn;
#pragma unroll
      for (int dt = 0; dt < 4; ++dt) O[dt][r] *= alpha;
      bf16* prow = &Pt[w][(hi * 8 + r) * LDH];
      prow[ln]      = f2bf(e0);
      prow[16 + ln] = f2bf(e1);
    }
    // reload P as A-fragment (C-layout -> A-layout via LDS), then O += P @ V
    Frag pF;
    {
      const bf16* pp = &Pt[w][ln * LDH + hi * 8];
      pF.h[0] = *(const v8bf*)pp;
      pF.h[1] = *(const v8bf*)(pp + 16);
    }
#pragma unroll
    for (int dt = 0; dt < 4; ++dt) {
      const bf16* pv = Vh + (size_t)(dt * 16 + ln) * LQ + kb + hi * 8;
      Frag vF;
      vF.h[0] = *(const v8bf*)pv;
      vF.h[1] = *(const v8bf*)(pv + 16);
      O[dt] = wmma_bf16(pF.v, vF.v, O[dt]);
    }
  }

  // finalize: O/l, store to (m=(l,b), c=h*64+d) bf16 buffer for output projection
#pragma unroll
  for (int r = 0; r < 8; ++r) {
    float inv = 1.0f / l_i[r];
    int q = q0 + hi * 8 + r;
    size_t baseo = ((size_t)q * BB + b) * CC + h * 64 + ln;
#pragma unroll
    for (int dt = 0; dt < 4; ++dt) Ob[baseo + dt * 16] = f2bf(O[dt][r] * inv);
  }
}

// ---------------------------------------------------------------- output projection
__global__ void __launch_bounds__(256)
outproj_kernel(const bf16* __restrict__ A, const bf16* __restrict__ W,
               const float* __restrict__ bias, float* __restrict__ out) {
  __shared__ __align__(16) bf16 As[2 * BUFH];
  __shared__ __align__(16) bf16 Bs[2 * BUFH];
  const int tid  = threadIdx.x;
  const int lane = tid & 31;
  const int wid  = tid >> 5;
  const int wm = wid & 3, wn = wid >> 2;
  const int m0 = blockIdx.y * BM;
  const int n0 = blockIdx.x * BN;
  const int ln = lane & 15;
  const int hi = lane >> 4;

  const unsigned ldsA = (unsigned)(size_t)As;
  const unsigned ldsB = (unsigned)(size_t)Bs;
  const unsigned long long gA = (unsigned long long)(size_t)(A + (size_t)m0 * KD);
  const unsigned long long gB = (unsigned long long)(size_t)(W + (size_t)n0 * KD);

  v8f acc[2][4];
#pragma unroll
  for (int i = 0; i < 2; ++i)
#pragma unroll
    for (int j = 0; j < 4; ++j) acc[i][j] = zero8();

  gemm_mainloop(As, Bs, ldsA, ldsB, gA, gB, wm, wn, ln, hi, tid, acc);

  const int rsel = hi * 8;
#pragma unroll
  for (int mt = 0; mt < 2; ++mt) {
#pragma unroll
    for (int nt = 0; nt < 4; ++nt) {
      int n = n0 + wn * 64 + nt * 16 + ln;
      float bv = bias[n];
#pragma unroll
      for (int r = 0; r < 8; ++r) {
        int m = m0 + wm * 32 + mt * 16 + rsel + r;
        out[(size_t)m * CC + n] = acc[mt][nt][r] + bv;
      }
    }
  }
}

// ---------------------------------------------------------------- launch
extern "C" void kernel_launch(void* const* d_in, const int* in_sizes, int n_in,
                              void* d_out, int out_size, void* d_ws, size_t ws_size,
                              hipStream_t stream) {
  (void)in_sizes; (void)n_in; (void)out_size; (void)ws_size;
  const float* hs   = (const float*)d_in[0];
  const float* rpe  = (const float*)d_in[1];
  const int*   mask = (const int*)d_in[2];
  const float* Wqkv = (const float*)d_in[3];
  const float* bqkv = (const float*)d_in[4];
  const float* Wo   = (const float*)d_in[5];
  const float* bo   = (const float*)d_in[6];
  float* out = (float*)d_out;

  const size_t SZ_HS = (size_t)MTOT * KD * 2;          // 16 MB
  const size_t SZ_WQ = (size_t)NQKV * KD * 2;          //  6 MB
  const size_t SZ_WO = (size_t)CC * KD * 2;            //  2 MB
  const size_t SZ_T  = (size_t)BB * HH * LQ * DD * 2;  // 16 MB (Q, K, Vt each)

  char* p = (char*)d_ws;
  bf16* hsb = (bf16*)p;  p += SZ_HS;
  bf16* wqb = (bf16*)p;  p += SZ_WQ;
  bf16* wob = (bf16*)p;  p += SZ_WO;
  bf16* Qb  = (bf16*)p;  p += SZ_T;
  bf16* Kb  = (bf16*)p;  p += SZ_T;
  bf16* Vt  = (bf16*)p;  p += SZ_T;
  bf16* Ob  = (bf16*)p;  p += SZ_T;   // 8192 x 1024 bf16

  const int n_hs = MTOT * KD, n_wq = NQKV * KD, n_wo = CC * KD;
  cvt_f32_bf16_kernel<<<n_hs / 256, 256, 0, stream>>>(hs,   hsb, n_hs);
  cvt_f32_bf16_kernel<<<n_wq / 256, 256, 0, stream>>>(Wqkv, wqb, n_wq);
  cvt_f32_bf16_kernel<<<n_wo / 256, 256, 0, stream>>>(Wo,   wob, n_wo);

  qkv_rope_kernel<<<dim3(NQKV / BN, MTOT / BM), 256, 0, stream>>>(hsb, wqb, bqkv, rpe, Qb, Kb, Vt);
  attn_kernel<<<BB * HH * (LQ / 64), 128, 0, stream>>>(Qb, Kb, Vt, mask, Ob);
  outproj_kernel<<<dim3(CC / BN, MTOT / BM), 256, 0, stream>>>(Ob, wob, bo, out);
}
